// GATClassifier_35287451304383
// MI455X (gfx1250) — matrix operand order repaired and checked by reading, not compiled, not run on hardware
//
#include <hip/hip_runtime.h>
#include <hip/hip_bf16.h>

// ---------------------------------------------------------------------------
// GAT classifier for MI455X (gfx1250, wave32).
// Dense projections: v_wmma_f32_16x16x32_f16, LDS-staged tiles with all
// fragment gathers as contiguous ds_load_b128 (weights pre-transposed so the
// B fragment is row-contiguous).  Edge softmax/aggregation: L2 float atomics.
// ---------------------------------------------------------------------------

typedef _Float16 f16;
typedef __attribute__((ext_vector_type(16))) _Float16 v16h;
typedef __attribute__((ext_vector_type(8)))  _Float16 v8h;
typedef __attribute__((ext_vector_type(8)))  float    v8f;

#define HEADS  8
#define HDIM   256
#define INDIM  512
#define OUTDIM 40

// ---------------------------------------------------------------- utilities
__global__ void k_f32_to_f16(const float* __restrict__ in, f16* __restrict__ out, int n) {
    for (int i = blockIdx.x * blockDim.x + threadIdx.x; i < n; i += gridDim.x * blockDim.x)
        out[i] = (f16)in[i];
}

// in: [K,N] f32 row-major  ->  out: [N,K] f16 (transposed, for WMMA B tiles)
__global__ void k_f32_to_f16_T(const float* __restrict__ in, f16* __restrict__ out,
                               int Kdim, int Ndim) {
    int i = blockIdx.x * blockDim.x + threadIdx.x;
    if (i >= Kdim * Ndim) return;
    const int n = i / Kdim, k = i - n * Kdim;
    out[i] = (f16)in[(size_t)k * Ndim + n];
}

__global__ void k_fill_f32(float* __restrict__ p, float v, int n) {
    for (int i = blockIdx.x * blockDim.x + threadIdx.x; i < n; i += gridDim.x * blockDim.x)
        p[i] = v;
}

__device__ inline float atomicMaxFloat(float* addr, float value) {
    // IEEE ordering trick; init value must be -inf (0xff800000).
    if (value >= 0.0f)
        return __int_as_float(atomicMax((int*)addr, __float_as_int(value)));
    else
        return __uint_as_float(atomicMin((unsigned int*)addr, __float_as_uint(value)));
}

// ------------------------------------------------------------- WMMA GEMM
// C[M,N] = A[M,K](f16) * B[K,N] (given as BT[N,K] f16) (+bias)(relu?)
//          -> f32 Cf and/or f16 Ch.
// Block: 256 threads = 8 wave32.  Block tile 64x64, K-step 64 (2 WMMA pairs
// per barrier).  Wave tile 16x32.  All LDS fragment reads are b128.
__global__ __launch_bounds__(256) void k_wmma_gemm(
    const f16* __restrict__ A, const f16* __restrict__ BT,
    const float* __restrict__ bias, float* __restrict__ Cf, f16* __restrict__ Ch,
    int M, int N, int K, int do_relu)
{
    __shared__ f16 As[64][72];   // 64x64 tile, pitch 144B (16B mult, 36-bank stride)
    __shared__ f16 Bs[64][72];   // BT tile: Bs[n][k]

    const int tid  = threadIdx.x;
    const int bm   = blockIdx.y * 64;
    const int bn   = blockIdx.x * 64;
    const int wave = tid >> 5;
    const int lane = tid & 31;
    const int m_base = (wave & 3) * 16;   // 4 wave-rows of 16
    const int n_base = (wave >> 2) * 32;  // 2 wave-cols of 32
    const int lmod   = lane & 15;
    const int lhalf  = lane >> 4;

    v8f acc0 = {};
    v8f acc1 = {};

    for (int k0 = 0; k0 < K; k0 += 64) {
        // ---- stage 64x64 A tile and 64x64 BT tile (uint4 per thread x4)
#pragma unroll
        for (int j = 0; j < 4; ++j) {
            const int c    = tid + j * 256;       // 0..1023
            const int row  = c >> 3;              // 0..127
            const int col8 = (c & 7) * 8;         // 0..56
            const int gm   = bm + row;
            uint4 va = {0u, 0u, 0u, 0u};
            if (gm < M) va = *reinterpret_cast<const uint4*>(A + (size_t)gm * K + k0 + col8);
            *reinterpret_cast<uint4*>(&As[row][col8]) = va;
            uint4 vb = *reinterpret_cast<const uint4*>(BT + (size_t)(bn + row) * K + k0 + col8);
            *reinterpret_cast<uint4*>(&Bs[row][col8]) = vb;
        }
        // prefetch next K tile (global_prefetch_b8)
        if (k0 + 64 < K) {
            const int row  = tid >> 3;
            const int col8 = (tid & 7) * 8;
            if (bm + row < M) __builtin_prefetch(A + (size_t)(bm + row) * K + k0 + 64 + col8, 0, 0);
            __builtin_prefetch(BT + (size_t)(bn + row) * K + k0 + 64 + col8, 0, 0);
        }
        __syncthreads();

        // ---- two K=32 WMMA steps; every fragment = two contiguous b128 loads
#pragma unroll
        for (int kk = 0; kk < 64; kk += 32) {
            // A 16x32 frag: lane(m,half) = A[m][half*16 .. +15] (contiguous in k, matches B)
            const v8h alo = *reinterpret_cast<const v8h*>(&As[m_base + lmod][kk + lhalf * 16]);
            const v8h ahi = *reinterpret_cast<const v8h*>(&As[m_base + lmod][kk + lhalf * 16 + 8]);
            const v16h af = __builtin_shufflevector(alo, ahi,
                0, 1, 2, 3, 4, 5, 6, 7, 8, 9, 10, 11, 12, 13, 14, 15);
            // B 32x16 frag: lane(n,half) = BT[n][half*16 .. +15] (contiguous in k)
            const v8h b0lo = *reinterpret_cast<const v8h*>(&Bs[n_base + lmod][kk + lhalf * 16]);
            const v8h b0hi = *reinterpret_cast<const v8h*>(&Bs[n_base + lmod][kk + lhalf * 16 + 8]);
            const v16h bf0 = __builtin_shufflevector(b0lo, b0hi,
                0, 1, 2, 3, 4, 5, 6, 7, 8, 9, 10, 11, 12, 13, 14, 15);
            const v8h b1lo = *reinterpret_cast<const v8h*>(&Bs[n_base + 16 + lmod][kk + lhalf * 16]);
            const v8h b1hi = *reinterpret_cast<const v8h*>(&Bs[n_base + 16 + lmod][kk + lhalf * 16 + 8]);
            const v16h bf1 = __builtin_shufflevector(b1lo, b1hi,
                0, 1, 2, 3, 4, 5, 6, 7, 8, 9, 10, 11, 12, 13, 14, 15);

            acc0 = __builtin_amdgcn_wmma_f32_16x16x32_f16(false, af, false, bf0,
                                                          (short)0, acc0, false, false);
            acc1 = __builtin_amdgcn_wmma_f32_16x16x32_f16(false, af, false, bf1,
                                                          (short)0, acc1, false, false);
        }
        __syncthreads();
    }

    // ---- epilogue: C 16x16 layout, VGPR r -> M = r + 8*lhalf, N = lmod
#pragma unroll
    for (int r = 0; r < 8; ++r) {
        const int row = bm + m_base + lhalf * 8 + r;
        if (row >= M) continue;
        const int c0 = bn + n_base + lmod;
        const int c1 = c0 + 16;
        float v0 = acc0[r], v1 = acc1[r];
        if (bias)    { v0 += bias[c0];      v1 += bias[c1]; }
        if (do_relu) { v0 = fmaxf(v0, 0.f); v1 = fmaxf(v1, 0.f); }
        if (Cf) { Cf[(size_t)row * N + c0] = v0;      Cf[(size_t)row * N + c1] = v1; }
        if (Ch) { Ch[(size_t)row * N + c0] = (f16)v0; Ch[(size_t)row * N + c1] = (f16)v1; }
    }
}

// ----------------------------------------------------- GAT edge pipeline
__global__ void k_node_alpha(const float* __restrict__ Wh,
                             const float* __restrict__ a_src,
                             const float* __restrict__ a_dst,
                             float* __restrict__ as, float* __restrict__ ad, int NH) {
    int i = blockIdx.x * blockDim.x + threadIdx.x;
    if (i >= NH) return;
    const int n = i >> 3, h = i & 7;
    const float* row = Wh + (size_t)n * HDIM + h * 32;
    const float* v1 = a_src + h * 32;
    const float* v2 = a_dst + h * 32;
    float s1 = 0.f, s2 = 0.f;
#pragma unroll
    for (int d = 0; d < 32; ++d) { float x = row[d]; s1 += x * v1[d]; s2 += x * v2[d]; }
    as[i] = s1;
    ad[i] = s2;
}

__global__ void k_edge_logits(const int* __restrict__ src, const int* __restrict__ dst,
                              const float* __restrict__ as, const float* __restrict__ ad,
                              float* __restrict__ pbuf, float* __restrict__ mmax, int E) {
    int e = blockIdx.x * blockDim.x + threadIdx.x;
    if (e >= E) return;
    const int s = src[e], d = dst[e];
#pragma unroll
    for (int h = 0; h < HEADS; ++h) {
        float x = as[s * HEADS + h] + ad[d * HEADS + h];
        x = x > 0.f ? x : 0.2f * x;
        pbuf[(size_t)e * HEADS + h] = x;
        atomicMaxFloat(&mmax[d * HEADS + h], x);
    }
}

__global__ void k_edge_exp(const int* __restrict__ dst, const float* __restrict__ mmax,
                           float* __restrict__ pbuf, float* __restrict__ denom, int E) {
    int e = blockIdx.x * blockDim.x + threadIdx.x;
    if (e >= E) return;
    const int d = dst[e];
#pragma unroll
    for (int h = 0; h < HEADS; ++h) {
        float v = __expf(pbuf[(size_t)e * HEADS + h] - mmax[d * HEADS + h]);
        pbuf[(size_t)e * HEADS + h] = v;
        atomicAdd(&denom[d * HEADS + h], v);
    }
}

// agg[dst, c] += attn * Wh[src, c]; one 256-thread block per edge, attn
// computed once per head and broadcast through LDS.
__global__ __launch_bounds__(256) void k_edge_scatter(
    const int* __restrict__ src, const int* __restrict__ dst,
    const float* __restrict__ pbuf, const float* __restrict__ denom,
    const float* __restrict__ Wh, float* __restrict__ agg, int E) {
    __shared__ float wsh[HEADS];
    const int e = blockIdx.x;
    if (e >= E) return;
    const int c = threadIdx.x;          // feature index 0..255
    const int s = src[e], d = dst[e];
    if (c < HEADS)
        wsh[c] = pbuf[(size_t)e * HEADS + c] / fmaxf(denom[d * HEADS + c], 1e-9f);
    __syncthreads();
    atomicAdd(&agg[(size_t)d * HDIM + c], wsh[c >> 5] * Wh[(size_t)s * HDIM + c]);
}

__global__ void k_relu_to_f16(const float* __restrict__ in, f16* __restrict__ out, int n) {
    for (int i = blockIdx.x * blockDim.x + threadIdx.x; i < n; i += gridDim.x * blockDim.x)
        out[i] = (f16)fmaxf(in[i], 0.f);
}

// logits = relu(h2) @ W_out + b_out; one block (64 thr) per node, row in LDS.
__global__ __launch_bounds__(64) void k_out_head(
    const float* __restrict__ h, const float* __restrict__ Wo,
    const float* __restrict__ bo, float* __restrict__ out, int Nn) {
    __shared__ float row[HDIM];
    const int n = blockIdx.x;
    if (n >= Nn) return;
    for (int i = threadIdx.x; i < HDIM; i += 64)
        row[i] = fmaxf(h[(size_t)n * HDIM + i], 0.f);   // relu folded into load
    __syncthreads();
    const int t = threadIdx.x;
    if (t < OUTDIM) {
        float acc = bo[t];
#pragma unroll 8
        for (int k = 0; k < HDIM; ++k) acc += row[k] * Wo[k * OUTDIM + t];
        out[(size_t)n * OUTDIM + t] = acc;
    }
}

// ---------------------------------------------------------------- driver
extern "C" void kernel_launch(void* const* d_in, const int* in_sizes, int n_in,
                              void* d_out, int out_size, void* d_ws, size_t ws_size,
                              hipStream_t stream) {
    const float* x    = (const float*)d_in[0];
    const int*   ei   = (const int*)d_in[1];
    const float* W_in = (const float*)d_in[2];
    const float* b_in = (const float*)d_in[3];
    const float* W1   = (const float*)d_in[4];
    const float* a1s  = (const float*)d_in[5];
    const float* a1d  = (const float*)d_in[6];
    const float* W2   = (const float*)d_in[7];
    const float* a2s  = (const float*)d_in[8];
    const float* a2d  = (const float*)d_in[9];
    const float* Wo   = (const float*)d_in[10];
    const float* bo   = (const float*)d_in[11];

    const int Nn = in_sizes[0] / INDIM;   // 10000
    const int E  = in_sizes[1] / 2;       // 320000
    const int* src = ei;
    const int* dst = ei + E;

    // ---- workspace layout (256 B aligned)
    char* ws = (char*)d_ws;
    size_t off = 0;
    auto take = [&](size_t bytes) { char* p = ws + off; off = (off + bytes + 255) & ~(size_t)255; return p; };
    f16*   F16A   = (f16*)take((size_t)Nn * INDIM * 2);       // x(f16), later h1(f16)
    f16*   F16B   = (f16*)take((size_t)Nn * HDIM * 2);        // h0(f16)
    f16*   WTin   = (f16*)take((size_t)INDIM * HDIM * 2);     // W_in^T [256][512]
    f16*   WT1    = (f16*)take((size_t)HDIM * HDIM * 2);      // W1^T   [256][256]
    f16*   WT2    = (f16*)take((size_t)HDIM * HDIM * 2);      // W2^T
    float* Wh     = (float*)take((size_t)Nn * HDIM * 4);      // per-layer projections
    float* agg    = (float*)take((size_t)Nn * HDIM * 4);      // aggregated messages
    float* as     = (float*)take((size_t)Nn * HEADS * 4);
    float* ad     = (float*)take((size_t)Nn * HEADS * 4);
    float* mmax   = (float*)take((size_t)Nn * HEADS * 4);
    float* denom  = (float*)take((size_t)Nn * HEADS * 4);
    float* pbuf   = (float*)take((size_t)E * HEADS * 4);
    (void)ws_size; (void)n_in; (void)out_size;

    const int T = 256;
    auto blks = [&](int n) { return (n + T - 1) / T; };

    // ---- f16 conversions (weights transposed for row-contiguous B fragments)
    k_f32_to_f16<<<blks(Nn * INDIM), T, 0, stream>>>(x, F16A, Nn * INDIM);
    k_f32_to_f16_T<<<blks(INDIM * HDIM), T, 0, stream>>>(W_in, WTin, INDIM, HDIM);
    k_f32_to_f16_T<<<blks(HDIM * HDIM), T, 0, stream>>>(W1, WT1, HDIM, HDIM);
    k_f32_to_f16_T<<<blks(HDIM * HDIM), T, 0, stream>>>(W2, WT2, HDIM, HDIM);

    const dim3 ggrid(HDIM / 64, (Nn + 63) / 64);

    // ---- input linear: h0 = x @ W_in + b_in  (f16 out only; feeds layer 1)
    k_wmma_gemm<<<ggrid, T, 0, stream>>>(F16A, WTin, b_in, nullptr, F16B,
                                         Nn, HDIM, INDIM, 0);

    // ---- GAT layer closure
    auto gat_layer = [&](const f16* hin16, const f16* WT,
                         const float* asrc, const float* adst) {
        k_wmma_gemm<<<ggrid, T, 0, stream>>>(hin16, WT, nullptr, Wh, nullptr,
                                             Nn, HDIM, HDIM, 0);
        k_node_alpha<<<blks(Nn * HEADS), T, 0, stream>>>(Wh, asrc, adst, as, ad, Nn * HEADS);
        k_fill_f32<<<blks(Nn * HEADS), T, 0, stream>>>(mmax, -INFINITY, Nn * HEADS);
        k_fill_f32<<<blks(Nn * HEADS), T, 0, stream>>>(denom, 0.f, Nn * HEADS);
        k_fill_f32<<<blks(Nn * HDIM), T, 0, stream>>>(agg, 0.f, Nn * HDIM);
        k_edge_logits<<<blks(E), T, 0, stream>>>(src, dst, as, ad, pbuf, mmax, E);
        k_edge_exp<<<blks(E), T, 0, stream>>>(dst, mmax, pbuf, denom, E);
        k_edge_scatter<<<E, T, 0, stream>>>(src, dst, pbuf, denom, Wh, agg, E);
    };

    // layer 1: h1 = relu(agg) -> f16 into F16A
    gat_layer(F16B, WT1, a1s, a1d);
    k_relu_to_f16<<<blks(Nn * HDIM), T, 0, stream>>>(agg, F16A, Nn * HDIM);

    // layer 2: relu folded into output head's LDS load
    gat_layer(F16A, WT2, a2s, a2d);

    // ---- output head
    k_out_head<<<Nn, 64, 0, stream>>>(agg, Wo, bo, (float*)d_out, Nn);
}
